// TransductiveAttentionProtocol_15006615733041
// MI455X (gfx1250) — matrix operand order repaired and checked by reading, not compiled, not run on hardware
//
#include <hip/hip_runtime.h>
#include <hip/hip_bf16.h>
#include <math.h>

typedef __attribute__((ext_vector_type(8)))  float  v8f;
typedef __attribute__((ext_vector_type(4)))  float  f4;
typedef __attribute__((ext_vector_type(16))) __bf16 v16bf;
typedef __attribute__((ext_vector_type(8)))  __bf16 v8bf;

#define WMMA_BF16(A, B, Cacc) \
  __builtin_amdgcn_wmma_f32_16x16x32_bf16(false, (A), false, (B), (short)0, (Cacc), false, false)

constexpr int CLS    = 64;         // classes
constexpr int D      = 128;        // emb dim
constexpr int F      = 512;        // in feat
constexpr int NS     = 3200;
constexpr int NQ     = 200000;
constexpr int NTOT   = NS + NQ;    // 203200 (multiple of 32 -> no padding)
constexpr int TILES  = NTOT / 16;  // 12700
constexpr int KT     = NTOT / 32;  // 6350
constexpr int QTILES = NQ / 16;    // 12500
constexpr int PUPD_CHUNK = 10;     // k-tiles per block in proto update (635*10 = 6350)

// ---------- helpers ----------
__device__ inline v16bf make_a_from_bf16(const __bf16* p) {
  // A fragment: 16B at p (K = +0..7) and 16B at p+16 (K = +16..23)
  v8bf lo = *(const v8bf*)p;
  v8bf hv = *(const v8bf*)(p + 16);
  v16bf a;
#pragma unroll
  for (int i = 0; i < 8; ++i) { a[i] = lo[i]; a[8 + i] = hv[i]; }
  return a;
}

// ---------- K0: pack W (f32 512x128) into bf16 B-operand layout ----------
__global__ void k_packW(const float* __restrict__ W, __bf16* __restrict__ Wb) {
  int tid = blockIdx.x * 256 + threadIdx.x;        // 65536 total
  int e  = tid & 15;
  int ln = (tid >> 4) & 31;
  int dt = (tid >> 9) & 7;
  int kt = tid >> 12;
  int k = kt * 32 + 8 * (ln >> 4) + (e & 7) + 16 * (e >> 3);
  int d = dt * 16 + (ln & 15);
  Wb[tid] = (__bf16)W[(size_t)k * D + d];
}

// ---------- K1: encoder GEMM emb = X@W + b, emit bf16 emb + row norms ----------
// W panel (128KB bf16) is staged once per block in LDS: B fragments come from
// the DS pipe and co-issue with the 410MB global A-stream (the HBM roofline).
__global__ void k_embed(const float* __restrict__ support, const float* __restrict__ query,
                        const __bf16* __restrict__ Wb, const float* __restrict__ bias,
                        __bf16* __restrict__ embRM, float* __restrict__ a2) {
  extern __shared__ __align__(32) char smemRaw[];
  __bf16* sWb = (__bf16*)smemRaw;                       // 65536 elems = 131072 B
  {
    const uint4* src = (const uint4*)Wb;
    uint4* dst = (uint4*)sWb;
    for (int i = threadIdx.x; i < (F * D * 2) / 16; i += 256) dst[i] = src[i];
  }
  __syncthreads();

  int w = threadIdx.x >> 5;
  int tile = blockIdx.x * 8 + w;
  if (tile >= TILES) return;
  int l = threadIdx.x & 31, hi = l >> 4, c16 = l & 15;

  int row = tile * 16 + c16;  // A-operand row for this lane
  const float* xrow = (row < NS) ? (support + (size_t)row * F)
                                 : (query + (size_t)(row - NS) * F);
  v8f acc[8];
#pragma unroll
  for (int nt = 0; nt < 8; ++nt) {
    float bb = bias[nt * 16 + c16];
#pragma unroll
    for (int i = 0; i < 8; ++i) acc[nt][i] = bb;
  }
  const v16bf* wb = (const v16bf*)sWb;   // LDS-resident B fragments
  for (int kk = 0; kk < F / 32; ++kk) {
    const f4* p = (const f4*)(xrow + kk * 32 + hi * 8);
    f4 f0 = p[0], f1 = p[1], f2 = p[4], f3 = p[5];
    v16bf a;
#pragma unroll
    for (int i = 0; i < 4; ++i) {
      a[i]      = (__bf16)f0[i];
      a[4 + i]  = (__bf16)f1[i];
      a[8 + i]  = (__bf16)f2[i];
      a[12 + i] = (__bf16)f3[i];
    }
#pragma unroll
    for (int nt = 0; nt < 8; ++nt)
      acc[nt] = WMMA_BF16(a, wb[(kk * 8 + nt) * 32 + l], acc[nt]);
  }
  // store bf16 embeddings (row-major) + row squared-norms
  float s[8];
#pragma unroll
  for (int v = 0; v < 8; ++v) s[v] = 0.f;
#pragma unroll
  for (int nt = 0; nt < 8; ++nt)
#pragma unroll
    for (int v = 0; v < 8; ++v) {
      float x = acc[nt][v];
      embRM[(size_t)(tile * 16 + v + 8 * hi) * D + nt * 16 + c16] = (__bf16)x;
      s[v] += x * x;
    }
#pragma unroll
  for (int m = 1; m < 16; m <<= 1)
#pragma unroll
    for (int v = 0; v < 8; ++v) s[v] += __shfl_xor(s[v], m, 32);
  float myv = 0.f;
#pragma unroll
  for (int v = 0; v < 8; ++v) if (c16 == v) myv = s[v];
  if (c16 < 8) a2[tile * 16 + 8 * hi + c16] = myv;
}

// ---------- K2: pack emb into B-operand layout ----------
__global__ void k_packEmbB(const __bf16* __restrict__ embRM, __bf16* __restrict__ embB) {
  size_t tid = (size_t)blockIdx.x * 256 + threadIdx.x;  // KT*8*32 threads
  int ln = (int)(tid & 31);
  int dt = (int)((tid >> 5) & 7);
  size_t kt = tid >> 8;
#pragma unroll
  for (int e = 0; e < 16; ++e)
    embB[tid * 16 + e] =
        embRM[((size_t)kt * 32 + 8 * (ln >> 4) + (e & 7) + 16 * (e >> 3)) * D + dt * 16 + (ln & 15)];
}

// ---------- proto zero / init ----------
__global__ void k_protoZero(float* __restrict__ proto, float* __restrict__ counts) {
  for (int i = threadIdx.x; i < CLS * D; i += 256) proto[i] = 0.f;
  if (threadIdx.x < CLS) counts[threadIdx.x] = 0.f;
}

__global__ void k_protoInit(const __bf16* __restrict__ embRM, const int* __restrict__ labels,
                            float* __restrict__ proto, float* __restrict__ counts) {
  int tid = blockIdx.x * 256 + threadIdx.x;  // NS*D
  int row = tid >> 7, d = tid & 127;
  int lab = labels[row];
  atomicAdd(&proto[(size_t)lab * D + d], (float)embRM[(size_t)row * D + d]);
  if (d == 0) atomicAdd(&counts[lab], 1.0f);
}

// ---------- pack prototypes (with scaling) into B layout + norms ----------
__global__ void k_packProto(const float* __restrict__ proto, const float* __restrict__ counts,
                            __bf16* __restrict__ Ppack, float* __restrict__ b2,
                            int mode, float invN) {
  int t = threadIdx.x;
  for (int o = t; o < CLS * D; o += 256) {
    int e  = o & 15;
    int ln = (o >> 4) & 31;
    int nt = (o >> 9) & 3;
    int kt = o >> 11;
    int n = nt * 16 + (ln & 15);
    int k = kt * 32 + 8 * (ln >> 4) + (e & 7) + 16 * (e >> 3);
    float sc = mode ? invN : 1.0f / fmaxf(counts[n], 1.0f);
    Ppack[o] = (__bf16)(proto[(size_t)n * D + k] * sc);
  }
  if (t < CLS) {
    float sc = mode ? invN : 1.0f / fmaxf(counts[t], 1.0f);
    float s = 0.f;
    for (int d = 0; d < D; ++d) { float x = proto[(size_t)t * D + d] * sc; s += x * x; }
    b2[t] = s;
  }
}

// ---------- K4: dist + softmax, store transposed attn (bf16 [64][NTOT]) ----------
__global__ void k_attn(const __bf16* __restrict__ embRM, const __bf16* __restrict__ Ppack,
                       const float* __restrict__ a2, const float* __restrict__ b2,
                       __bf16* __restrict__ attnT) {
  __shared__ __align__(32) __bf16 sm[8][CLS * 16];   // 16 KB transpose staging
  __shared__ __align__(32) __bf16 sP[CLS * D];       // 16 KB prototype panel
  {
    const uint4* src = (const uint4*)Ppack;
    uint4* dst = (uint4*)sP;
    for (int i = threadIdx.x; i < (CLS * D * 2) / 16; i += 256) dst[i] = src[i];
  }
  __syncthreads();

  int w = threadIdx.x >> 5;
  int tile = blockIdx.x * 8 + w;
  if (tile >= TILES) return;
  int l = threadIdx.x & 31, hi = l >> 4, c16 = l & 15;

  v8f acc[4];
#pragma unroll
  for (int nt = 0; nt < 4; ++nt)
#pragma unroll
    for (int i = 0; i < 8; ++i) acc[nt][i] = 0.f;

  const __bf16* arow = embRM + (size_t)(tile * 16 + c16) * D;
  const v16bf* pb = (const v16bf*)sP;
#pragma unroll
  for (int kk = 0; kk < 4; ++kk) {
    v16bf a = make_a_from_bf16(arow + kk * 32 + hi * 8);
#pragma unroll
    for (int nt = 0; nt < 4; ++nt)
      acc[nt] = WMMA_BF16(a, pb[(kk * 4 + nt) * 32 + l], acc[nt]);
  }
  float b2c[4];
#pragma unroll
  for (int nt = 0; nt < 4; ++nt) b2c[nt] = b2[nt * 16 + c16];

#pragma unroll
  for (int v = 0; v < 8; ++v) {
    int row = tile * 16 + v + 8 * hi;
    float aa = a2[row];
    float vals[4];
#pragma unroll
    for (int nt = 0; nt < 4; ++nt)
      vals[nt] = -sqrtf(fmaxf(aa + b2c[nt] - 2.0f * acc[nt][v], 0.0f));
    float m = fmaxf(fmaxf(vals[0], vals[1]), fmaxf(vals[2], vals[3]));
#pragma unroll
    for (int msk = 1; msk < 16; msk <<= 1) m = fmaxf(m, __shfl_xor(m, msk, 32));
    float e[4], s = 0.f;
#pragma unroll
    for (int nt = 0; nt < 4; ++nt) { e[nt] = expf(vals[nt] - m); s += e[nt]; }
#pragma unroll
    for (int msk = 1; msk < 16; msk <<= 1) s += __shfl_xor(s, msk, 32);
    float inv = 1.0f / s;
#pragma unroll
    for (int nt = 0; nt < 4; ++nt)
      sm[w][(nt * 16 + c16) * 16 + v + 8 * hi] = (__bf16)(e[nt] * inv);
  }
  // wave-private LDS transpose -> contiguous 32B stores per class row
#pragma unroll
  for (int g = 0; g < 2; ++g) {
    int cc = l + g * 32;
    v16bf tv = *(const v16bf*)&sm[w][cc * 16];
    *(v16bf*)(attnT + (size_t)cc * NTOT + tile * 16) = tv;
  }
}

// ---------- K5: prototypes = attnT @ embB (K split over blocks, atomic reduce) ----------
__global__ void k_protoUpd(const __bf16* __restrict__ attnT, const __bf16* __restrict__ embB,
                           float* __restrict__ proto) {
  int w = threadIdx.x >> 5, l = threadIdx.x & 31, hi = l >> 4, c16 = l & 15;
  int mt = w & 3;            // class tile 0..3
  int nb = (w >> 2) * 4;     // emb-dim tile base (0 or 4)
  v8f acc[4];
#pragma unroll
  for (int j = 0; j < 4; ++j)
#pragma unroll
    for (int i = 0; i < 8; ++i) acc[j][i] = 0.f;

  const v16bf* eb = (const v16bf*)embB;
  for (int kt = 0; kt < PUPD_CHUNK; ++kt) {
    int kg = blockIdx.x * PUPD_CHUNK + kt;
    v16bf a = make_a_from_bf16(attnT + (size_t)(mt * 16 + c16) * NTOT + kg * 32 + hi * 8);
#pragma unroll
    for (int j = 0; j < 4; ++j)
      acc[j] = WMMA_BF16(a, eb[(size_t)(kg * 8 + nb + j) * 32 + l], acc[j]);
  }
#pragma unroll
  for (int j = 0; j < 4; ++j)
#pragma unroll
    for (int v = 0; v < 8; ++v)
      atomicAdd(&proto[(size_t)(mt * 16 + v + 8 * hi) * D + (nb + j) * 16 + c16], acc[j][v]);
}

// ---------- K6: logits = -dist(q_emb, protos) ----------
__global__ void k_logits(const __bf16* __restrict__ embRM, const __bf16* __restrict__ Ppack,
                         const float* __restrict__ a2, const float* __restrict__ b2,
                         float* __restrict__ out) {
  __shared__ __align__(32) __bf16 sP[CLS * D];       // 16 KB prototype panel
  {
    const uint4* src = (const uint4*)Ppack;
    uint4* dst = (uint4*)sP;
    for (int i = threadIdx.x; i < (CLS * D * 2) / 16; i += 256) dst[i] = src[i];
  }
  __syncthreads();

  int w = threadIdx.x >> 5;
  int tile = blockIdx.x * 8 + w;
  if (tile >= QTILES) return;
  int l = threadIdx.x & 31, hi = l >> 4, c16 = l & 15;

  v8f acc[4];
#pragma unroll
  for (int nt = 0; nt < 4; ++nt)
#pragma unroll
    for (int i = 0; i < 8; ++i) acc[nt][i] = 0.f;

  const __bf16* arow = embRM + (size_t)(NS + tile * 16 + c16) * D;
  const v16bf* pb = (const v16bf*)sP;
#pragma unroll
  for (int kk = 0; kk < 4; ++kk) {
    v16bf a = make_a_from_bf16(arow + kk * 32 + hi * 8);
#pragma unroll
    for (int nt = 0; nt < 4; ++nt)
      acc[nt] = WMMA_BF16(a, pb[(kk * 4 + nt) * 32 + l], acc[nt]);
  }
  float b2c[4];
#pragma unroll
  for (int nt = 0; nt < 4; ++nt) b2c[nt] = b2[nt * 16 + c16];
#pragma unroll
  for (int v = 0; v < 8; ++v) {
    float aa = a2[NS + tile * 16 + v + 8 * hi];
#pragma unroll
    for (int nt = 0; nt < 4; ++nt) {
      float val = -sqrtf(fmaxf(aa + b2c[nt] - 2.0f * acc[nt][v], 0.0f));
      out[(size_t)(tile * 16 + v + 8 * hi) * CLS + nt * 16 + c16] = val;
    }
  }
}

// ---------- host ----------
extern "C" void kernel_launch(void* const* d_in, const int* in_sizes, int n_in,
                              void* d_out, int out_size, void* d_ws, size_t ws_size,
                              hipStream_t stream) {
  (void)in_sizes; (void)n_in; (void)out_size; (void)ws_size;
  const float* support = (const float*)d_in[0];
  const float* query   = (const float*)d_in[1];
  const int*   labels  = (const int*)d_in[2];
  const float* W       = (const float*)d_in[3];
  const float* bias    = (const float*)d_in[4];
  float* out = (float*)d_out;

  char* p = (char*)d_ws;
  __bf16* Wb     = (__bf16*)p; p += (size_t)F * D * 2;          // 128 KB
  __bf16* embRM  = (__bf16*)p; p += (size_t)NTOT * D * 2;       // 52 MB
  __bf16* embB   = (__bf16*)p; p += (size_t)NTOT * D * 2;       // 52 MB
  float*  a2     = (float*)p;  p += (size_t)NTOT * 4;           // 0.8 MB
  __bf16* attnT  = (__bf16*)p; p += (size_t)CLS * NTOT * 2;     // 26 MB
  float*  proto  = (float*)p;  p += (size_t)CLS * D * 4;        // 32 KB
  float*  counts = (float*)p;  p += 256;
  __bf16* Ppack  = (__bf16*)p; p += (size_t)CLS * D * 2;        // 16 KB
  float*  b2     = (float*)p;  p += 256;

  k_packW<<<(F * D) / 256, 256, 0, stream>>>(W, Wb);
  k_embed<<<(TILES + 7) / 8, 256, (size_t)F * D * 2, stream>>>(support, query, Wb, bias,
                                                               embRM, a2);
  k_packEmbB<<<(KT * 8 * 32) / 256, 256, 0, stream>>>(embRM, embB);
  k_protoZero<<<1, 256, 0, stream>>>(proto, counts);
  k_protoInit<<<(NS * D) / 256, 256, 0, stream>>>(embRM, labels, proto, counts);
  k_packProto<<<1, 256, 0, stream>>>(proto, counts, Ppack, b2, 0, 0.0f);

  const float invN = 1.0f / (float)NTOT;
  for (int it = 0; it < 3; ++it) {
    k_attn<<<(TILES + 7) / 8, 256, 0, stream>>>(embRM, Ppack, a2, b2, attnT);
    k_protoZero<<<1, 256, 0, stream>>>(proto, counts);
    k_protoUpd<<<KT / PUPD_CHUNK, 256, 0, stream>>>(attnT, embB, proto);
    k_packProto<<<1, 256, 0, stream>>>(proto, counts, Ppack, b2, 1, invN);
  }
  k_logits<<<(QTILES + 7) / 8, 256, 0, stream>>>(embRM, Ppack, a2, b2, out);
}
